// MultiHeadAttention_68899865362474
// MI455X (gfx1250) — compile-verified, matrix operand order checked by reading, not run on hardware
//
#include <hip/hip_runtime.h>

#define Bc 8
#define Tc 1024
#define Ec 1024
#define Hc 16
#define Dc 64
#define SCLD 1056   // LDS row stride (bf16 elems), multiple of 32 -> 32B-aligned frags

typedef __attribute__((ext_vector_type(16))) __bf16         bf16x16;
typedef __attribute__((ext_vector_type(8)))  float          f32x8;
typedef __attribute__((ext_vector_type(16))) unsigned short u16x16;
typedef __attribute__((ext_vector_type(8)))  unsigned short u16x8;
typedef __attribute__((ext_vector_type(4)))  int            i32x4;

#define AS1 __attribute__((address_space(1)))
#define AS3 __attribute__((address_space(3)))

#if defined(__has_builtin)
#if __has_builtin(__builtin_amdgcn_global_load_async_to_lds_b128)
#define HAVE_ASYNC_LDS 1
#endif
#endif

// swap bit3 <-> bit4 of an index: maps fragment k-order {0-7,16-23,8-15,24-31}
// to contiguous storage within each 32-wide k group. Bijection; higher bits kept.
__device__ __forceinline__ int swz(int c) {
  return (c & ~24) | ((c & 8) << 1) | ((c & 16) >> 1);
}

__device__ __forceinline__ unsigned short f2b(float x) {
  __bf16 h = (__bf16)x;
  return __builtin_bit_cast(unsigned short, h);
}
__device__ __forceinline__ float b2f(unsigned short u) {
  return __builtin_bit_cast(float, (unsigned int)u << 16);
}

// One contiguous 32-byte fragment load (storage is k-swizzled).
__device__ __forceinline__ bf16x16 fragld(const unsigned short* p) {
  return __builtin_bit_cast(bf16x16, *(const u16x16*)p);
}

#define WMMA_BF16(a, b, c) \
  __builtin_amdgcn_wmma_f32_16x16x32_bf16(false, (a), false, (b), (short)0, (c), false, false)

// --- 64B global->LDS row copy: async (ASYNCcnt) if available, sync fallback ---
__device__ __forceinline__ void stage_row64(const unsigned short* g, unsigned short* l) {
#ifdef HAVE_ASYNC_LDS
#pragma unroll
  for (int c = 0; c < 4; ++c)
    __builtin_amdgcn_global_load_async_to_lds_b128(
        (AS1 i32x4*)(unsigned long long)(uintptr_t)(g + c * 8),
        (AS3 i32x4*)(unsigned int)(uintptr_t)(l + c * 8), 0, 0);
#else
#pragma unroll
  for (int c = 0; c < 4; ++c)
    *(u16x8*)(l + c * 8) = *(const u16x8*)(g + c * 8);
#endif
}

__device__ __forceinline__ void stage_wait() {
#ifdef HAVE_ASYNC_LDS
#if __has_builtin(__builtin_amdgcn_s_wait_asynccnt)
  __builtin_amdgcn_s_wait_asynccnt(0);
#else
  asm volatile("s_wait_asynccnt 0x0" ::: "memory");
#endif
#endif
}

// ---------------------------------------------------------------------------
// fp32 -> bf16 convert with k-swizzle (8 elems / thread).
// ---------------------------------------------------------------------------
__global__ void cvt_swz_kernel(const float* __restrict__ src,
                               unsigned short* __restrict__ dst, int n8) {
  int i = blockIdx.x * blockDim.x + threadIdx.x;
  if (i >= n8) return;
  long long e  = (long long)i * 8;
  long long ed = (e & ~24LL) | ((e & 8LL) << 1) | ((e & 16LL) >> 1);
  const float4* p = (const float4*)(src + e);
  float4 a = p[0], b = p[1];
  u16x8 o;
  o[0] = f2b(a.x); o[1] = f2b(a.y); o[2] = f2b(a.z); o[3] = f2b(a.w);
  o[4] = f2b(b.x); o[5] = f2b(b.y); o[6] = f2b(b.z); o[7] = f2b(b.w);
  *(u16x8*)(dst + ed) = o;
}

// ---------------------------------------------------------------------------
// GEMM mainloop shared by both projection kernels:
// block 128 threads (4 waves), wave tile 16(M) x 128(N), K in 32-chunks.
// B tile (128x32) double-buffered in LDS via async copy; A streamed direct.
// ---------------------------------------------------------------------------
__device__ __forceinline__ void gemm_mainloop(const unsigned short* __restrict__ arow,
                                              const unsigned short* __restrict__ W,
                                              int n0, int hl16, int mrow,
                                              unsigned short* btile /* [2][128*32] */,
                                              f32x8 acc[8]) {
  const int tid = threadIdx.x;                    // 0..127 -> one B row each
  const unsigned short* bsrc = W + (long long)(n0 + tid) * Ec;

  stage_row64(bsrc + 0, btile + tid * 32);
  stage_wait();
  __syncthreads();

  for (int kk = 0; kk < Ec; kk += 32) {
    const int cur = (kk >> 5) & 1;
    unsigned short* bt = btile + cur * (128 * 32);
    if (kk + 32 < Ec)
      stage_row64(bsrc + kk + 32, btile + (cur ^ 1) * (128 * 32) + tid * 32);
    __builtin_prefetch(arow + kk + 256, 0, 1);    // global_prefetch_b8 for A stream
    bf16x16 a = fragld(arow + kk + hl16);
#pragma unroll
    for (int t = 0; t < 8; ++t)
      acc[t] = WMMA_BF16(a, fragld(bt + (t * 16 + mrow) * 32 + hl16), acc[t]);
    if (kk + 32 < Ec) stage_wait();
    __syncthreads();
  }
}

// vmode 0: Y head-major [b,h][t][swz(d)]  (Q, K)
// vmode 1: Y transposed [b,h][d][swz(t)]  (V -> feeds P*V B-fragments)
__global__ void proj_gemm_kernel(const unsigned short* __restrict__ X,
                                 const unsigned short* __restrict__ W,
                                 unsigned short* __restrict__ Y,
                                 int vmode) {
  __shared__ __align__(64) unsigned short btile[2 * 128 * 32];
  const int lane = threadIdx.x & 31;
  const int wave = threadIdx.x >> 5;
  const int mrow = lane & 15;
  const int hl16 = (lane >> 4) * 16;
  const int m0   = blockIdx.y * 64 + wave * 16;
  const int n0   = blockIdx.x * 128;

  f32x8 acc[8] = {};
  gemm_mainloop(X + (long long)(m0 + mrow) * Ec, W, n0, hl16, mrow, btile, acc);

  const int hl8 = hl16 >> 1;  // 0 or 8
#pragma unroll
  for (int t = 0; t < 8; ++t) {
#pragma unroll
    for (int r = 0; r < 8; ++r) {
      int gm = m0 + r + hl8;             // C layout: vgpr r -> M=r / M=r+8
      int gn = n0 + t * 16 + mrow;       //           N = lane&15
      int b  = gm >> 10, tt = gm & (Tc - 1);
      int h  = gn >> 6,  d  = gn & (Dc - 1);
      long long idx = vmode
          ? ((long long)((b * Hc + h) * Dc + d) * Tc + swz(tt))
          : ((long long)((b * Hc + h) * Tc + tt) * Dc + swz(d));
      Y[idx] = f2b(acc[t][r]);
    }
  }
}

// ---------------------------------------------------------------------------
// Causal attention: one wave per (b, h, 16-query tile). Raw scores written
// directly as bf16 into the P buffer in fragment-storage order; softmax is
// order-independent so the passes walk storage order and exp() overwrites
// in place. 1/sum folded into the output scale. ~34 KB LDS per block.
// ---------------------------------------------------------------------------
__global__ void attn_kernel(const unsigned short* __restrict__ Q,
                            const unsigned short* __restrict__ K,
                            const unsigned short* __restrict__ Vt,
                            unsigned short* __restrict__ O) {
  __shared__ __align__(64) unsigned short sp[16 * SCLD];  // scores -> P (bf16)
  __shared__ float srinv[16];

  const int lane = threadIdx.x & 31;
  const int mrow = lane & 15;
  const int hl   = lane >> 4;
  const int hl16 = hl * 16;
  const int q0   = blockIdx.x * 16;
  const int h    = blockIdx.y;
  const int b    = blockIdx.z;
  const int nk   = q0 + 16;               // causal: only k <= q needed
  const int nk32 = (nk + 31) & ~31;

  const unsigned short* Qbh = Q  + (long long)((b * Hc + h) * Tc) * Dc;
  const unsigned short* Kbh = K  + (long long)((b * Hc + h) * Tc) * Dc;
  const unsigned short* Vbh = Vt + (long long)((b * Hc + h) * Dc) * Tc;

  bf16x16 aq0 = fragld(Qbh + (q0 + mrow) * Dc + 0  + hl16);
  bf16x16 aq1 = fragld(Qbh + (q0 + mrow) * Dc + 32 + hl16);

  const unsigned short NEGBF = f2b(-1e30f);

  for (int k0 = 0; k0 <= q0; k0 += 16) {
    bf16x16 bk0 = fragld(Kbh + (k0 + mrow) * Dc + 0  + hl16);
    bf16x16 bk1 = fragld(Kbh + (k0 + mrow) * Dc + 32 + hl16);
    f32x8 s = {};
    s = WMMA_BF16(aq0, bk0, s);
    s = WMMA_BF16(aq1, bk1, s);
    const int js = swz(k0 + mrow);
#pragma unroll
    for (int r = 0; r < 8; ++r) {
      int m = r + 8 * hl;
      float v = s[r] * 0.125f;                       // 1/sqrt(64)
      if (k0 + mrow > q0 + m) v = -1e30f;            // bites only on diagonal tile
      sp[m * SCLD + js] = f2b(v);
    }
  }
  if (nk32 != nk) {                                  // phantom columns -> -inf
#pragma unroll
    for (int c = 0; c < 8; ++c)
      sp[mrow * SCLD + swz(nk + 2 * c + hl)] = NEGBF;
  }
  __syncthreads();

  // softmax over each row's nk32 storage slots; (row = lane&15, half = lane>>4)
  {
    const int row = mrow;
    float mx = -1e30f;
    for (int j = hl; j < nk32; j += 2) mx = fmaxf(mx, b2f(sp[row * SCLD + j]));
    mx = fmaxf(mx, __shfl_xor(mx, 16, 32));
    float sum = 0.f;
    for (int j = hl; j < nk32; j += 2) {
      float e = __expf(b2f(sp[row * SCLD + j]) - mx);
      sum += e;
      sp[row * SCLD + j] = f2b(e);                   // in-place: same element
    }
    sum += __shfl_xor(sum, 16, 32);
    if (lane < 16) srinv[row] = 1.0f / sum;
  }
  __syncthreads();

  // O = P * V
  f32x8 c0 = {}, c1 = {}, c2 = {}, c3 = {};
  for (int k0 = 0; k0 < nk32; k0 += 32) {
    bf16x16 ap = fragld(sp + mrow * SCLD + k0 + hl16);
    c0 = WMMA_BF16(ap, fragld(Vbh + (0 * 16 + mrow) * Tc + k0 + hl16), c0);
    c1 = WMMA_BF16(ap, fragld(Vbh + (1 * 16 + mrow) * Tc + k0 + hl16), c1);
    c2 = WMMA_BF16(ap, fragld(Vbh + (2 * 16 + mrow) * Tc + k0 + hl16), c2);
    c3 = WMMA_BF16(ap, fragld(Vbh + (3 * 16 + mrow) * Tc + k0 + hl16), c3);
  }
  f32x8 cc[4] = {c0, c1, c2, c3};
#pragma unroll
  for (int t = 0; t < 4; ++t) {
#pragma unroll
    for (int r = 0; r < 8; ++r) {
      int m = r + 8 * hl;
      float v = cc[t][r] * srinv[m];
      int col = h * Dc + t * 16 + mrow;
      // concat-heads [B*T, E], k-swizzled for the out-projection A fragments
      O[(long long)(b * Tc + q0 + m) * Ec + swz(col)] = f2b(v);
    }
  }
}

// ---------------------------------------------------------------------------
// Output projection: out = A(bf16, swizzled) @ Wfc.T + bfc, f32 output.
// ---------------------------------------------------------------------------
__global__ void outproj_kernel(const unsigned short* __restrict__ A,
                               const unsigned short* __restrict__ W,
                               const float* __restrict__ bias,
                               float* __restrict__ out) {
  __shared__ __align__(64) unsigned short btile[2 * 128 * 32];
  const int lane = threadIdx.x & 31;
  const int wave = threadIdx.x >> 5;
  const int mrow = lane & 15;
  const int hl16 = (lane >> 4) * 16;
  const int m0   = blockIdx.y * 64 + wave * 16;
  const int n0   = blockIdx.x * 128;

  f32x8 acc[8] = {};
  gemm_mainloop(A + (long long)(m0 + mrow) * Ec, W, n0, hl16, mrow, btile, acc);

  const int hl8 = hl16 >> 1;
  float bv[8];
#pragma unroll
  for (int t = 0; t < 8; ++t) bv[t] = bias[n0 + t * 16 + mrow];
#pragma unroll
  for (int t = 0; t < 8; ++t) {
#pragma unroll
    for (int r = 0; r < 8; ++r) {
      int gm = m0 + r + hl8;
      int gn = n0 + t * 16 + mrow;
      out[(long long)gm * Ec + gn] = acc[t][r] + bv[t];
    }
  }
}

extern "C" void kernel_launch(void* const* d_in, const int* in_sizes, int n_in,
                              void* d_out, int out_size, void* d_ws, size_t ws_size,
                              hipStream_t stream) {
  (void)in_sizes; (void)n_in; (void)out_size; (void)ws_size;
  const float* values  = (const float*)d_in[0];
  const float* keys    = (const float*)d_in[1];
  const float* queries = (const float*)d_in[2];
  // d_in[3] = mask (causality implemented analytically)
  const float* Wq  = (const float*)d_in[4];
  const float* Wk  = (const float*)d_in[5];
  const float* Wv  = (const float*)d_in[6];
  const float* Wfc = (const float*)d_in[7];
  const float* bfc = (const float*)d_in[8];

  const size_t NE = (size_t)Bc * Tc * Ec;   // 8M elements (inputs / activations)
  const size_t NW = (size_t)Ec * Ec;        // 1M elements (weights)
  unsigned short* xq  = (unsigned short*)d_ws;   // swizzled bf16 copies
  unsigned short* xk  = xq  + NE;
  unsigned short* xv  = xk  + NE;
  unsigned short* wq  = xv  + NE;
  unsigned short* wk  = wq  + NW;
  unsigned short* wv  = wk  + NW;
  unsigned short* wfc = wv  + NW;
  unsigned short* qbuf  = wfc  + NW;   // [b,h][t][swz(d)]
  unsigned short* kbuf  = qbuf + NE;   // [b,h][t][swz(d)]
  unsigned short* vtbuf = kbuf + NE;   // [b,h][d][swz(t)]
  unsigned short* abuf  = vtbuf + NE;  // [B*T][swz(e)]

  const int cb = 256;
  cvt_swz_kernel<<<(int)(NE / 8 / cb), cb, 0, stream>>>(queries, xq, (int)(NE / 8));
  cvt_swz_kernel<<<(int)(NE / 8 / cb), cb, 0, stream>>>(keys,    xk, (int)(NE / 8));
  cvt_swz_kernel<<<(int)(NE / 8 / cb), cb, 0, stream>>>(values,  xv, (int)(NE / 8));
  cvt_swz_kernel<<<(int)(NW / 8 / cb), cb, 0, stream>>>(Wq,  wq,  (int)(NW / 8));
  cvt_swz_kernel<<<(int)(NW / 8 / cb), cb, 0, stream>>>(Wk,  wk,  (int)(NW / 8));
  cvt_swz_kernel<<<(int)(NW / 8 / cb), cb, 0, stream>>>(Wv,  wv,  (int)(NW / 8));
  cvt_swz_kernel<<<(int)(NW / 8 / cb), cb, 0, stream>>>(Wfc, wfc, (int)(NW / 8));

  dim3 gg(Ec / 128, (Bc * Tc) / 64);                   // (8, 128)
  proj_gemm_kernel<<<gg, 128, 0, stream>>>(xq, wq, qbuf, 0);
  proj_gemm_kernel<<<gg, 128, 0, stream>>>(xk, wk, kbuf, 0);
  proj_gemm_kernel<<<gg, 128, 0, stream>>>(xv, wv, vtbuf, 1);

  attn_kernel<<<dim3(Tc / 16, Hc, Bc), 32, 0, stream>>>(qbuf, kbuf, vtbuf, abuf);

  outproj_kernel<<<gg, 128, 0, stream>>>(abuf, wfc, bfc, (float*)d_out);
}
